// GravityDisplacement_48782238548328
// MI455X (gfx1250) — matrix operand order repaired and checked by reading, not compiled, not run on hardware
//
#include <hip/hip_runtime.h>
#include <hip/hip_bf16.h>
#include <math.h>

#define BT 4
#define LQ 2048
#define DD 256
#define PHN 64
#define EHN 256
#define K1 320
#define H2 128
#define NR (BT*LQ)   // 8192 total rows

static constexpr float MAX_DISP = 5.0f;
static constexpr float MIN_DISP = 0.5f;
static constexpr float REPW     = 0.3f;
static constexpr float SURFACE  = 103.0f;
static constexpr float SPACING  = SURFACE / 45.0f;
static constexpr float DANGER   = SPACING / 4.0f;
static constexpr float EOFF     = 0.1f;

typedef __attribute__((ext_vector_type(16))) _Float16 v16h;
typedef __attribute__((ext_vector_type(8)))  float    v8f;
typedef unsigned int v4u __attribute__((ext_vector_type(4)));
typedef int          v8i __attribute__((ext_vector_type(8)));
typedef int          v4i __attribute__((ext_vector_type(4)));

#if defined(__has_builtin)
#  if __has_builtin(__builtin_amdgcn_tensor_load_to_lds)
#    define HAVE_TDM 1
#  endif
#endif
#if defined(__has_include)
#  if __has_include(<hip/amd_detail/amd_gfx1250_TDM.h>)
#    define TDM_SIX_ARGS 1
#  endif
#endif

union AF { v16h h; uint4 u[2]; };

__device__ __forceinline__ float gelu_exact(float x) {
  return 0.5f * x * (1.0f + erff(x * 0.70710678118654752f));
}
__device__ __forceinline__ float softplus_f(float x) {
  return fmaxf(x, 0.0f) + log1pf(expf(-fabsf(x)));
}

// ---------------------------------------------------------------------------
// Repack weights: Wt1[n][k] = em_w1[k][n] as f16 (K-contiguous per column),
//                 Wt2[n][k] = em_w2[k][n] as f16.
// ---------------------------------------------------------------------------
__global__ void k_prep_weights(const float* __restrict__ w1, const float* __restrict__ w2,
                               _Float16* __restrict__ Wt1, _Float16* __restrict__ Wt2) {
  int idx = blockIdx.x * 256 + threadIdx.x;
  if (idx < K1 * EHN) {
    int n = idx / K1, k = idx % K1;
    Wt1[idx] = (_Float16)w1[k * EHN + n];
  } else {
    int j = idx - K1 * EHN;
    if (j < EHN * H2) {
      int n = j / EHN, k = j % EHN;
      Wt2[j] = (_Float16)w2[k * H2 + n];
    }
  }
}

// ---------------------------------------------------------------------------
// Positional-encoder MLP (2->64 gelu ->64) + build f16 X = [latents | pf].
// One 64-thread block per row.
// ---------------------------------------------------------------------------
__global__ __launch_bounds__(64) void k_prep_x(
    const float* __restrict__ latents, const float* __restrict__ positions,
    const float* __restrict__ pw1, const float* __restrict__ pb1,
    const float* __restrict__ pw2, const float* __restrict__ pb2,
    _Float16* __restrict__ X) {
  __shared__ float sh1[PHN];
  const int r = blockIdx.x;
  const int t = threadIdx.x;
  const float px = positions[r * 2 + 0];
  const float py = positions[r * 2 + 1];
  float h = px * pw1[t] + py * pw1[PHN + t] + pb1[t];
  sh1[t] = gelu_exact(h);
  __syncthreads();
  float acc = pb2[t];
#pragma unroll 8
  for (int k = 0; k < PHN; ++k) acc += sh1[k] * pw2[k * PHN + t];
  _Float16* xr = X + (size_t)r * K1;
  xr[DD + t] = (_Float16)acc;  // no activation on second pe layer
#pragma unroll
  for (int q = 0; q < 4; ++q) {
    int c = t * 4 + q;
    xr[c] = (_Float16)latents[(size_t)r * DD + c];
  }
}

// ---------------------------------------------------------------------------
// Fused error-MLP: GEMM1(320->256, WMMA f16) + LayerNorm + GELU +
//                  GEMM2(256->128, WMMA f16) + GELU + GEMV(w3) + softplus.
// 256 threads = 8 waves; block tile = 64 rows. Waves: 4 row-tiles x 2 col-grps.
// X tile is staged into LDS by the Tensor Data Mover (one issue from wave 0).
// ---------------------------------------------------------------------------
__global__ __launch_bounds__(256) void k_mlp(
    const _Float16* __restrict__ X, const _Float16* __restrict__ Wt1,
    const _Float16* __restrict__ Wt2, const float* __restrict__ b1,
    const float* __restrict__ lng, const float* __restrict__ lnb,
    const float* __restrict__ b2, const float* __restrict__ w3,
    const float* __restrict__ b3, float* __restrict__ errors) {
  __shared__ __align__(16) _Float16 sT[64 * K1];        // X tile, reused as H tile
  __shared__ float sSum[64][2], sSq[64][2], sEr[64][2];

  const int tid  = threadIdx.x;
  const int row0 = blockIdx.x * 64;

  __builtin_prefetch(Wt1, 0, 1);
  __builtin_prefetch(Wt2, 0, 1);

#if defined(HAVE_TDM)
  // --- TDM: DMA the 64x320 f16 tile (one 2D tensor tile) into LDS ---------
  if (tid < 32) {  // wave 0 issues the descriptor; TDM ignores EXEC
    unsigned long long ga = (unsigned long long)(uintptr_t)(X + (size_t)row0 * K1);
    unsigned ldsoff = (unsigned)(uintptr_t)&sT[0];      // LDS byte offset = addr[31:0]
    v4u g0;
    g0[0] = 1u;                                          // count=1, user descriptor
    g0[1] = ldsoff;                                      // lds_addr
    g0[2] = (unsigned)(ga & 0xFFFFFFFFu);                // global_addr[31:0]
    g0[3] = (unsigned)((ga >> 32) & 0x01FFFFFFu) | (2u << 30);  // addr[56:32] | type=2
    v8i g1;
    g1[0] = (int)(1u << 16);        // workgroup_mask=0, data_size=1 (2 bytes)
    g1[1] = (int)(320u << 16);      // tensor_dim0[15:0]=320 (bits 63:48)
    g1[2] = (int)(8192u << 16);     // tensor_dim0 hi=0, tensor_dim1[15:0]=8192
    g1[3] = (int)(320u << 16);      // tensor_dim1 hi=0, tile_dim0=320
    g1[4] = 64;                     // tile_dim1=64, tile_dim2=0
    g1[5] = 320;                    // tensor_dim0_stride[31:0]=320
    g1[6] = 0;                      // stride hi / dim1 stride lo
    g1[7] = 0;
    v4i z = {0, 0, 0, 0};
#if defined(TDM_SIX_ARGS)
    v8i z8 = {0, 0, 0, 0, 0, 0, 0, 0};
    __builtin_amdgcn_tensor_load_to_lds(g0, g1, z, z, z8, 0);
#else
    __builtin_amdgcn_tensor_load_to_lds(g0, g1, z, z, 0);
#endif
    __builtin_amdgcn_s_wait_tensorcnt(0);
  }
  __syncthreads();
#else
  // --- fallback: cooperative b128 staging ---------------------------------
  {
    const uint4* src = (const uint4*)(X + (size_t)row0 * K1);
    uint4* dst = (uint4*)sT;
#pragma unroll
    for (int i = 0; i < 10; ++i) dst[tid + i * 256] = src[tid + i * 256];
  }
  __syncthreads();
#endif

  const int wave = tid >> 5, lane = tid & 31;
  const int ln = lane & 15, hi = lane >> 4;
  const int wr = (wave & 3) * 16;   // wave row-tile base
  const int cg = wave >> 2;         // column group (0/1)
  const int wc = cg * 128;          // GEMM1 col base
  const int arow = wr + ln;

  // ---------------- GEMM1: H = X @ W1 + b1 ----------------
  v8f acc[8];
#pragma unroll
  for (int t = 0; t < 8; ++t) {
    float bv = b1[wc + t * 16 + ln];
#pragma unroll
    for (int v = 0; v < 8; ++v) acc[t][v] = bv;
  }
  for (int kk = 0; kk < 10; ++kk) {
    const int k = kk * 32;
    AF a;
    a.u[0] = *(const uint4*)&sT[arow * K1 + k + hi * 8];
    a.u[1] = *(const uint4*)&sT[arow * K1 + k + hi * 8 + 16];
#pragma unroll
    for (int t = 0; t < 8; ++t) {
      const _Float16* bp = Wt1 + (wc + t * 16 + ln) * K1 + k + hi * 8;
      AF bm;
      bm.u[0] = *(const uint4*)bp;
      bm.u[1] = *(const uint4*)(bp + 16);
      acc[t] = __builtin_amdgcn_wmma_f32_16x16x32_f16(
          false, a.h, false, bm.h, (short)0, acc[t], false, false);
    }
  }

  // ---------------- LayerNorm (cross-wave) ----------------
  float s1[8], s2[8];
#pragma unroll
  for (int v = 0; v < 8; ++v) { s1[v] = 0.f; s2[v] = 0.f; }
#pragma unroll
  for (int t = 0; t < 8; ++t)
#pragma unroll
    for (int v = 0; v < 8; ++v) { float x = acc[t][v]; s1[v] += x; s2[v] += x * x; }
#pragma unroll
  for (int m = 1; m < 16; m <<= 1) {
#pragma unroll
    for (int v = 0; v < 8; ++v) {
      s1[v] += __shfl_xor(s1[v], m, 32);
      s2[v] += __shfl_xor(s2[v], m, 32);
    }
  }
  if (ln == 0) {
#pragma unroll
    for (int v = 0; v < 8; ++v) {
      int row = wr + v + hi * 8;
      sSum[row][cg] = s1[v];
      sSq[row][cg]  = s2[v];
    }
  }
  __syncthreads();
  float mu[8], rstd[8];
#pragma unroll
  for (int v = 0; v < 8; ++v) {
    int row = wr + v + hi * 8;
    float m_ = (sSum[row][0] + sSum[row][1]) * (1.0f / EHN);
    float q_ = (sSq[row][0] + sSq[row][1]) * (1.0f / EHN);
    mu[v] = m_;
    rstd[v] = rsqrtf(q_ - m_ * m_ + 1e-5f);
  }

  // normalize + gelu, write f16 H tile into sT (all X reads are done)
  _Float16* sH = sT;
#pragma unroll
  for (int t = 0; t < 8; ++t) {
    int n = wc + t * 16 + ln;
    float g_ = lng[n], bb = lnb[n];
#pragma unroll
    for (int v = 0; v < 8; ++v) {
      int row = wr + v + hi * 8;
      float x = (acc[t][v] - mu[v]) * rstd[v] * g_ + bb;
      sH[row * EHN + n] = (_Float16)gelu_exact(x);
    }
  }
  __syncthreads();

  // ---------------- GEMM2: G = gelu(H @ W2 + b2) ----------------
  const int wc2 = cg * 64;
  v8f a2[4];
#pragma unroll
  for (int t = 0; t < 4; ++t) {
    float bv = b2[wc2 + t * 16 + ln];
#pragma unroll
    for (int v = 0; v < 8; ++v) a2[t][v] = bv;
  }
  for (int kk = 0; kk < 8; ++kk) {
    const int k = kk * 32;
    AF a;
    a.u[0] = *(const uint4*)&sH[arow * EHN + k + hi * 8];
    a.u[1] = *(const uint4*)&sH[arow * EHN + k + hi * 8 + 16];
#pragma unroll
    for (int t = 0; t < 4; ++t) {
      const _Float16* bp = Wt2 + (wc2 + t * 16 + ln) * EHN + k + hi * 8;
      AF bm;
      bm.u[0] = *(const uint4*)bp;
      bm.u[1] = *(const uint4*)(bp + 16);
      a2[t] = __builtin_amdgcn_wmma_f32_16x16x32_f16(
          false, a.h, false, bm.h, (short)0, a2[t], false, false);
    }
  }

  // ---------------- GEMV with w3 + softplus ----------------
  float p[8];
#pragma unroll
  for (int v = 0; v < 8; ++v) p[v] = 0.f;
#pragma unroll
  for (int t = 0; t < 4; ++t) {
    float w = w3[wc2 + t * 16 + ln];
#pragma unroll
    for (int v = 0; v < 8; ++v) p[v] += gelu_exact(a2[t][v]) * w;
  }
#pragma unroll
  for (int m = 1; m < 16; m <<= 1)
#pragma unroll
    for (int v = 0; v < 8; ++v) p[v] += __shfl_xor(p[v], m, 32);
  if (ln == 0) {
#pragma unroll
    for (int v = 0; v < 8; ++v) sEr[wr + v + hi * 8][cg] = p[v];
  }
  __syncthreads();
  if (tid < 64) {
    float e = sEr[tid][0] + sEr[tid][1] + b3[0];
    errors[row0 + tid] = softplus_f(e);
  }
}

// ---------------------------------------------------------------------------
// Per-batch min / max of errors (deterministic tree reduce).
// emm layout: [emin x4][emax x4]
// ---------------------------------------------------------------------------
__global__ __launch_bounds__(256) void k_minmax(const float* __restrict__ errors,
                                                float* __restrict__ emm) {
  __shared__ float smn[256], smx[256];
  const int b = blockIdx.x, tid = threadIdx.x;
  float mn = 1e30f, mx = -1e30f;
  for (int i = tid; i < LQ; i += 256) {
    float e = errors[b * LQ + i];
    mn = fminf(mn, e);
    mx = fmaxf(mx, e);
  }
  smn[tid] = mn; smx[tid] = mx;
  __syncthreads();
  for (int s = 128; s > 0; s >>= 1) {
    if (tid < s) {
      smn[tid] = fminf(smn[tid], smn[tid + s]);
      smx[tid] = fmaxf(smx[tid], smx[tid + s]);
    }
    __syncthreads();
  }
  if (tid == 0) { emm[b] = smn[0]; emm[4 + b] = smx[0]; }
}

// ---------------------------------------------------------------------------
// All-pairs gravity + repulsion, LDS-tiled N-body. One thread per i-point.
// ---------------------------------------------------------------------------
__global__ __launch_bounds__(256) void k_forces(const float* __restrict__ positions,
                                                const float* __restrict__ errors,
                                                const float* __restrict__ emm,
                                                float* __restrict__ tf) {
  __shared__ float sx[256], sy[256], se[256];
  const int b  = blockIdx.x >> 3;
  const int i  = ((blockIdx.x & 7) << 8) + threadIdx.x;
  const int gi = b * LQ + i;
  const float emin = emm[b], emax = emm[4 + b];
  const float escale = 1.0f / (emax - emin + 1e-8f);
  const float xi = positions[gi * 2 + 0];
  const float yi = positions[gi * 2 + 1];
  float gx = 0.f, gy = 0.f, rx = 0.f, ry = 0.f;
  for (int jt = 0; jt < LQ; jt += 256) {
    const int gj = b * LQ + jt + threadIdx.x;
    sx[threadIdx.x] = positions[gj * 2 + 0];
    sy[threadIdx.x] = positions[gj * 2 + 1];
    se[threadIdx.x] = (errors[gj] - emin) * escale + EOFF;
    __syncthreads();
#pragma unroll 4
    for (int jj = 0; jj < 256; ++jj) {
      if (jt + jj == i) continue;  // exclude self
      float dx = sx[jj] - xi, dy = sy[jj] - yi;
      float d2 = dx * dx + dy * dy;
      float dist = fmaxf(sqrtf(d2), 1e-6f);
      float inv = 1.0f / dist;
      float dirx = dx * inv, diry = dy * inv;
      float gmag = se[jj] * inv * inv;            // POWER = 2
      gx += dirx * gmag; gy += diry * gmag;
      float viol = fmaxf(DANGER - dist, 0.0f);
      float rmag = expf(viol * (1.0f / DANGER)) - 1.0f;  // exact 0 when viol==0
      rx += dirx * rmag; ry += diry * rmag;
    }
    __syncthreads();
  }
  // total = gravity + REP * (-(sum dir*rmag))
  tf[gi * 2 + 0] = gx - REPW * rx;
  tf[gi * 2 + 1] = gy - REPW * ry;
}

// ---------------------------------------------------------------------------
// Per-batch sum of |force| (deterministic).
// ---------------------------------------------------------------------------
__global__ __launch_bounds__(256) void k_fsum(const float* __restrict__ tf,
                                              float* __restrict__ fsum) {
  __shared__ float ss[256];
  const int b = blockIdx.x, tid = threadIdx.x;
  float s = 0.f;
  for (int i = tid; i < LQ; i += 256) {
    int gi = b * LQ + i;
    float tx = tf[gi * 2], ty = tf[gi * 2 + 1];
    s += sqrtf(tx * tx + ty * ty);
  }
  ss[tid] = s;
  __syncthreads();
  for (int st = 128; st > 0; st >>= 1) {
    if (tid < st) ss[tid] += ss[tid + st];
    __syncthreads();
  }
  if (tid == 0) fsum[b] = ss[0];
}

// ---------------------------------------------------------------------------
// Finalize: displacement clamp + outputs [new_pos | displacement | errors].
// ---------------------------------------------------------------------------
__global__ __launch_bounds__(256) void k_final(const float* __restrict__ positions,
                                               const float* __restrict__ tf,
                                               const float* __restrict__ fsum,
                                               const float* __restrict__ errors,
                                               float* __restrict__ out) {
  const int gi = blockIdx.x * 256 + threadIdx.x;
  const int b  = gi / LQ;
  float tx = tf[gi * 2], ty = tf[gi * 2 + 1];
  float fmag  = sqrtf(tx * tx + ty * ty);
  float fmean = fsum[b] * (1.0f / LQ);
  float invf  = 1.0f / (fmag + 1e-8f);
  float frel  = fmag / (fmean + 1e-8f);
  float dmag  = fminf(fmaxf(frel * MAX_DISP, MIN_DISP), MAX_DISP);
  float dxo = tx * invf * dmag, dyo = ty * invf * dmag;
  out[gi * 2 + 0] = positions[gi * 2 + 0] + dxo;
  out[gi * 2 + 1] = positions[gi * 2 + 1] + dyo;
  out[NR * 2 + gi * 2 + 0] = dxo;
  out[NR * 2 + gi * 2 + 1] = dyo;
  out[NR * 4 + gi] = errors[gi];
}

// ---------------------------------------------------------------------------
extern "C" void kernel_launch(void* const* d_in, const int* in_sizes, int n_in,
                              void* d_out, int out_size, void* d_ws, size_t ws_size,
                              hipStream_t stream) {
  const float* latents   = (const float*)d_in[0];
  const float* positions = (const float*)d_in[1];
  const float* pe_w1 = (const float*)d_in[2];
  const float* pe_b1 = (const float*)d_in[3];
  const float* pe_w2 = (const float*)d_in[4];
  const float* pe_b2 = (const float*)d_in[5];
  const float* em_w1 = (const float*)d_in[6];
  const float* em_b1 = (const float*)d_in[7];
  const float* ln_g  = (const float*)d_in[8];
  const float* ln_b  = (const float*)d_in[9];
  const float* em_w2 = (const float*)d_in[10];
  const float* em_b2 = (const float*)d_in[11];
  const float* em_w3 = (const float*)d_in[12];
  const float* em_b3 = (const float*)d_in[13];
  // input 14 = layer_idx (unused: shared weights)

  char* ws = (char*)d_ws;
  _Float16* X    = (_Float16*)(ws);                 // 8192*320*2 = 5,242,880
  _Float16* Wt1  = (_Float16*)(ws + 5242880);       // 256*320*2  =   163,840
  _Float16* Wt2  = (_Float16*)(ws + 5406720);       // 128*256*2  =    65,536
  float*    errs = (float*)   (ws + 5472256);       // 8192*4     =    32,768
  float*    emm  = (float*)   (ws + 5505024);       // 8 floats
  float*    fsum = (float*)   (ws + 5505024 + 64);  // 4 floats
  float*    tf   = (float*)   (ws + 5505024 + 128); // 8192*2*4   =    65,536
  float*    out  = (float*)d_out;

  hipLaunchKernelGGL(k_prep_weights, dim3(448), dim3(256), 0, stream, em_w1, em_w2, Wt1, Wt2);
  hipLaunchKernelGGL(k_prep_x, dim3(NR), dim3(64), 0, stream,
                     latents, positions, pe_w1, pe_b1, pe_w2, pe_b2, X);
  hipLaunchKernelGGL(k_mlp, dim3(NR / 64), dim3(256), 0, stream,
                     X, Wt1, Wt2, em_b1, ln_g, ln_b, em_b2, em_w3, em_b3, errs);
  hipLaunchKernelGGL(k_minmax, dim3(4), dim3(256), 0, stream, errs, emm);
  hipLaunchKernelGGL(k_forces, dim3(32), dim3(256), 0, stream, positions, errs, emm, tf);
  hipLaunchKernelGGL(k_fsum, dim3(4), dim3(256), 0, stream, tf, fsum);
  hipLaunchKernelGGL(k_final, dim3(32), dim3(256), 0, stream, positions, tf, fsum, errs, out);
}
